// ScaledDotProductAttention_45165876085421
// MI455X (gfx1250) — compile-verified, matrix operand order checked by reading, not run on hardware
//
#include <hip/hip_runtime.h>
#include <stdint.h>

typedef __attribute__((ext_vector_type(16))) _Float16 v16h;
typedef __attribute__((ext_vector_type(8)))  float    v8f;

#define B_    4
#define H_    16
#define L_    2048
#define D_    64
#define TILE  16
#define NT    (L_ / TILE)     // 128 column tiles
#define NW    4               // waves per block
#define NCH   (L_ / 32)       // 64 V chunks of 32 rows
#define SCOL  (L_ + 8)        // padded score row stride (f16)
#define V32P  68              // padded f32 V-staging row stride

// Build 8 consecutive f16 fragment elements from 8 consecutive f32 in memory.
#define LOAD8_F16(frag, ebase, ptr)                                   \
    do {                                                              \
        float4 _x = *(const float4*)(ptr);                            \
        float4 _y = *(const float4*)((ptr) + 4);                      \
        (frag)[(ebase) + 0] = (_Float16)_x.x;                         \
        (frag)[(ebase) + 1] = (_Float16)_x.y;                         \
        (frag)[(ebase) + 2] = (_Float16)_x.z;                         \
        (frag)[(ebase) + 3] = (_Float16)_x.w;                         \
        (frag)[(ebase) + 4] = (_Float16)_y.x;                         \
        (frag)[(ebase) + 5] = (_Float16)_y.y;                         \
        (frag)[(ebase) + 6] = (_Float16)_y.z;                         \
        (frag)[(ebase) + 7] = (_Float16)_y.w;                         \
    } while (0)

__launch_bounds__(128, 1)
__global__ void sdpa_wmma_kernel(const float* __restrict__ Q,
                                 const float* __restrict__ K,
                                 const float* __restrict__ V,
                                 const unsigned char* __restrict__ mask,
                                 float* __restrict__ res,
                                 float* __restrict__ attn)
{
    __shared__ _Float16 sc[TILE][SCOL];             // scores, then exp(s-max)
    __shared__ float    vt[NW][2][32][V32P];        // per-wave V double buffer (async dest)
    __shared__ float    obuf[TILE][D_];             // O accumulation
    __shared__ float    pmax[TILE][8], psum[TILE][8];
    __shared__ float    smax[TILE],    ssum[TILE];

    const int tid = threadIdx.x;
    const int wid = tid >> 5;
    const int lid = tid & 31;
    const int g   = lid >> 4;       // half-wave group (selects +8 K-offset)
    const int ln  = lid & 15;       // row (A) / col (B,D) index
    const int d0  = 8 * g;          // base d-offset of this lane's fragment run

    const int blk   = blockIdx.x;
    const int itile = blk & (NT - 1);
    const int bh    = blk >> 7;               // b*H + h
    const int b     = bh >> 4;

    const size_t rowBase = (size_t)bh * L_ * D_ + (size_t)itile * TILE * D_;
    const float* Qg = Q + rowBase;
    const float* Kg = K + (size_t)bh * L_ * D_;
    const float* Vg = V + (size_t)bh * L_ * D_;
    const unsigned char* Mg = mask + (size_t)b * L_ * L_ + (size_t)itile * TILE * L_;
    float* attnBase = attn + (size_t)bh * L_ * L_ + (size_t)itile * TILE * L_;
    float* resBase  = res + rowBase;

    // zero O accumulator
    for (int e = tid; e < TILE * D_; e += 128)
        ((float*)obuf)[e] = 0.0f;

    // ---- A fragments for Q: built directly from global (contiguous k-map runs) ----
    // 16-bit A layout (ISA 7.12.2): lane ln holds row M=ln; K = 8g + {0..7}, 16+8g+{0..7}, ...
    v16h aq0, aq1;
    {
        const float* qrow = Qg + (size_t)ln * D_;
        LOAD8_F16(aq0, 0, qrow + d0);
        LOAD8_F16(aq0, 8, qrow + 16 + d0);
        LOAD8_F16(aq1, 0, qrow + 32 + d0);
        LOAD8_F16(aq1, 8, qrow + 48 + d0);
    }

    // ---- Pass 1: S = Q K^T / 8, mask, store f16 scores to LDS ----
    // B fragment B[k][n] = K[row n][k]: same contiguous runs -> direct global loads.
    for (int j = wid; j < NT; j += NW) {
        const float* krow = Kg + (size_t)(j * TILE + ln) * D_;
        v16h bk0, bk1;
        LOAD8_F16(bk0, 0, krow + d0);
        LOAD8_F16(bk0, 8, krow + 16 + d0);
        LOAD8_F16(bk1, 0, krow + 32 + d0);
        LOAD8_F16(bk1, 8, krow + 48 + d0);

        v8f c = {};
        c = __builtin_amdgcn_wmma_f32_16x16x32_f16(false, aq0, false, bk0,
                                                   (short)0, c, false, false);
        c = __builtin_amdgcn_wmma_f32_16x16x32_f16(false, aq1, false, bk1,
                                                   (short)0, c, false, false);
        // D layout: element r -> row r+8g, col = j*16+ln
        #pragma unroll
        for (int r = 0; r < 8; ++r) {
            const int row = r + 8 * g;
            const int col = j * TILE + ln;
            float sv = c[r] * 0.125f;                       // / TEMPERATURE
            if (Mg[(size_t)row * L_ + col]) sv = -60000.0f; // f16-finite mask fill
            sc[row][col] = (_Float16)sv;
        }
    }
    __syncthreads();

    // ---- Softmax: row max ----
    {
        const int row = tid >> 3, seg = tid & 7;
        float m = -3.4e38f;
        for (int cx = seg * 256; cx < seg * 256 + 256; ++cx)
            m = fmaxf(m, (float)sc[row][cx]);
        pmax[row][seg] = m;
    }
    __syncthreads();
    if (tid < TILE) {
        float m = pmax[tid][0];
        #pragma unroll
        for (int s = 1; s < 8; ++s) m = fmaxf(m, pmax[tid][s]);
        smax[tid] = m;
    }
    __syncthreads();

    // ---- exp(s - max): overwrite LDS (values in [0,1] -> f16 safe), row sums ----
    {
        const int row = tid >> 3, seg = tid & 7;
        const float m = smax[row];
        float sum = 0.0f;
        for (int cx = seg * 256; cx < seg * 256 + 256; ++cx) {
            float ev = __expf((float)sc[row][cx] - m);
            sum += ev;
            sc[row][cx] = (_Float16)ev;
        }
        psum[row][seg] = sum;
    }
    __syncthreads();
    if (tid < TILE) {
        float s = 0.0f;
        #pragma unroll
        for (int k = 0; k < 8; ++k) s += psum[tid][k];
        ssum[tid] = s;
    }
    __syncthreads();

    // ---- stream normalized probabilities to global (the 1 GiB write) ----
    for (int row = 0; row < TILE; ++row) {
        const float inv = 1.0f / ssum[row];
        float* arow = attnBase + (size_t)row * L_;
        #pragma unroll
        for (int q = 0; q < 4; ++q) {
            const int col = q * 512 + tid * 4;
            float4 o;
            o.x = (float)sc[row][col + 0] * inv;
            o.y = (float)sc[row][col + 1] * inv;
            o.z = (float)sc[row][col + 2] * inv;
            o.w = (float)sc[row][col + 3] * inv;
            *(float4*)(arow + col) = o;
        }
    }

    // ---- Pass 2: O = P V, V chunks double-buffered via async global->LDS copies ----
    // Each async b128 copies 16B/lane straight into LDS (no VGPR round trip),
    // tracked by ASYNCcnt; in-order completion lets s_wait_asynccnt 16 release
    // the previous group while the next streams in.
    auto issueV = [&](int ch, int bufIdx) {
        const int mbase = ch * 32;
        const unsigned ldsBase = (unsigned)(uintptr_t)(&vt[wid][bufIdx][0][0]);
        #pragma unroll
        for (int q = 0; q < 16; ++q) {
            const int idx = q * 128 + lid * 4;
            const int r = idx >> 6, d = idx & 63;
            const unsigned lofs = ldsBase + (unsigned)((r * V32P + d) * sizeof(float));
            const unsigned gofs = (unsigned)(((mbase + r) * D_ + d) * sizeof(float));
            asm volatile("global_load_async_to_lds_b128 %0, %1, %2"
                         :
                         : "v"(lofs), "v"(gofs), "s"(Vg)
                         : "memory");
        }
    };

    v8f acc0 = {}, acc1 = {}, acc2 = {}, acc3 = {};
    issueV(wid, 0);                       // prefetch first chunk
    int buf = 0;
    for (int ch = wid; ch < NCH; ch += NW) {
        const int nch = ch + NW;
        if (nch < NCH) {
            issueV(nch, buf ^ 1);         // stream next chunk into other buffer
            asm volatile("s_wait_asynccnt 16" ::: "memory");  // current chunk landed
        } else {
            asm volatile("s_wait_asynccnt 0" ::: "memory");
        }
        const int mbase = ch * 32;

        // A fragment: P[ln][mbase + k] from the f16 exp strip (contiguous runs)
        v16h ap;
        #pragma unroll
        for (int e = 0; e < 16; ++e) {
            const int v = e >> 1, s = e & 1;
            const int k = d0 + ((v < 4) ? (2 * v + s) : (16 + 2 * (v - 4) + s));
            ap[e] = sc[ln][mbase + k];
        }
        // B fragments: B[k][n] = V[mbase+k][t*16+ln] from async-staged f32
        #pragma unroll
        for (int t = 0; t < 4; ++t) {
            v16h bv;
            #pragma unroll
            for (int e = 0; e < 16; ++e) {
                const int v = e >> 1, s = e & 1;
                const int k = d0 + ((v < 4) ? (2 * v + s) : (16 + 2 * (v - 4) + s));
                bv[e] = (_Float16)vt[wid][buf][k][t * 16 + ln];
            }
            if      (t == 0) acc0 = __builtin_amdgcn_wmma_f32_16x16x32_f16(false, ap, false, bv, (short)0, acc0, false, false);
            else if (t == 1) acc1 = __builtin_amdgcn_wmma_f32_16x16x32_f16(false, ap, false, bv, (short)0, acc1, false, false);
            else if (t == 2) acc2 = __builtin_amdgcn_wmma_f32_16x16x32_f16(false, ap, false, bv, (short)0, acc2, false, false);
            else             acc3 = __builtin_amdgcn_wmma_f32_16x16x32_f16(false, ap, false, bv, (short)0, acc3, false, false);
        }
        buf ^= 1;
    }

    // cross-wave reduction via LDS float atomics (ds_add_f32)
    #pragma unroll
    for (int r = 0; r < 8; ++r) {
        const int row = r + 8 * g;
        atomicAdd(&obuf[row][ 0 + ln], acc0[r]);
        atomicAdd(&obuf[row][16 + ln], acc1[r]);
        atomicAdd(&obuf[row][32 + ln], acc2[r]);
        atomicAdd(&obuf[row][48 + ln], acc3[r]);
    }
    __syncthreads();

    // final: scale by 1/rowsum and write res (coalesced)
    #pragma unroll
    for (int q = 0; q < 8; ++q) {
        const int e = tid + q * 128;
        const int row = e >> 6, d = e & 63;
        resBase[(size_t)row * D_ + d] = obuf[row][d] * (1.0f / ssum[row]);
    }
}

extern "C" void kernel_launch(void* const* d_in, const int* in_sizes, int n_in,
                              void* d_out, int out_size, void* d_ws, size_t ws_size,
                              hipStream_t stream) {
    (void)in_sizes; (void)n_in; (void)d_ws; (void)ws_size; (void)out_size;
    const float* Q = (const float*)d_in[0];
    const float* K = (const float*)d_in[1];
    const float* V = (const float*)d_in[2];
    const unsigned char* mask = (const unsigned char*)d_in[3];
    float* res  = (float*)d_out;
    float* attn = res + (size_t)B_ * H_ * L_ * D_;   // outputs concatenated flat

    dim3 grid(B_ * H_ * NT);   // 8192 blocks: one 16-row Q tile each
    dim3 block(128);           // 4 waves (wave32)
    sdpa_wmma_kernel<<<grid, block, 0, stream>>>(Q, K, V, mask, res, attn);
}
